// GATBlock_34711925686357
// MI455X (gfx1250) — matrix operand order repaired and checked by reading, not compiled, not run on hardware
//
#include <hip/hip_runtime.h>

typedef _Float16 half_t;
typedef __attribute__((ext_vector_type(16))) _Float16 v16h;
typedef __attribute__((ext_vector_type(8)))  _Float16 v8h;
typedef __attribute__((ext_vector_type(8)))  float    v8f;
typedef __attribute__((ext_vector_type(4)))  int      v4i;

#define AS1 __attribute__((address_space(1)))
#define AS3 __attribute__((address_space(3)))

#if defined(__has_builtin)
#if __has_builtin(__builtin_amdgcn_global_load_async_to_lds_b128)
#define HAVE_ASYNC_LDS 1
#endif
#endif

__device__ __forceinline__ void wait_async0() {
#if defined(__has_builtin) && __has_builtin(__builtin_amdgcn_s_wait_asynccnt)
    __builtin_amdgcn_s_wait_asynccnt(0);
#else
    asm volatile("s_wait_asynccnt 0x0" ::: "memory");
#endif
}

// ---------- order-preserving float <-> uint encoding (for atomicMax softmax max) ----------
__device__ __forceinline__ unsigned fenc(float f) {
    unsigned u = __float_as_uint(f);
    return (u & 0x80000000u) ? ~u : (u | 0x80000000u);
}
__device__ __forceinline__ float fdec(unsigned u) {
    unsigned b = (u & 0x80000000u) ? (u ^ 0x80000000u) : ~u;
    return __uint_as_float(b);
}

// ---------- 1. per-head edge coefficient k[h] = sum_c W_edge[h*16+c]*att_edge[h,c] ----------
__global__ void k_edge_coef(const float* __restrict__ W_edge,
                            const float* __restrict__ att_edge,
                            float* __restrict__ kE) {
    int h = threadIdx.x;
    if (h < 8) {
        float s = 0.f;
        #pragma unroll
        for (int c = 0; c < 16; ++c) s += W_edge[h * 16 + c] * att_edge[h * 16 + c];
        kE[h] = s;
    }
}

// ---------- 2. incoming-edge stats for self-loop attr (fill_value='mean') ----------
__global__ void k_loop_stats(const int* __restrict__ ei, const float* __restrict__ ea,
                             unsigned* __restrict__ cnt, float* __restrict__ ssum, int E) {
    int e = blockIdx.x * blockDim.x + threadIdx.x;
    if (e >= E) return;
    int d = ei[E + e];                       // edge_index[1][e]
    atomicAdd(&cnt[d], 1u);
    atomicAdd(&ssum[d], ea[e]);
}

__global__ void k_loop_attr(const unsigned* __restrict__ cnt, const float* __restrict__ ssum,
                            float* __restrict__ la, int N) {
    int n = blockIdx.x * blockDim.x + threadIdx.x;
    if (n >= N) return;
    unsigned c = cnt[n];
    la[n] = c ? (ssum[n] / (float)c) : 0.f;
}

// ---------- 3. fp32 -> fp16 node features ----------
__global__ void k_convert_x(const float* __restrict__ x, half_t* __restrict__ xh, int total) {
    int i = blockIdx.x * blockDim.x + threadIdx.x;
    if (i < total) xh[i] = (half_t)x[i];
}

// ---------- 4. pack W into WMMA B-fragment lane order ----------
// B tile (kc,ct): b[e] of lane = Wt[kc*32 + kb + e][ct*16 + (lane&15)],  kb = (lane&16)?16:0
// Wt[k][d] = W[d][k]  (xp = x @ W^T)
__global__ void k_pack_w(const float* __restrict__ W, half_t* __restrict__ Bp) {
    int idx = blockIdx.x * blockDim.x + threadIdx.x;   // 8*4*32*16 = 16384 packed halves
    if (idx >= 16384) return;
    int e    = idx & 15;
    int lane = (idx >> 4) & 31;
    int kc   = (idx >> 9) & 3;
    int ct   = (idx >> 11) & 7;
    int dcol = ct * 16 + (lane & 15);
    int k    = kc * 32 + ((lane & 16) ? 16 : 0) + e;
    Bp[idx]  = (half_t)W[dcol * 128 + k];
}

// ---------- 5. WMMA projection: xp = x @ W^T, fused a_src/a_dst head reductions ----------
// block = 16 nodes x 128 cols; wave ct owns head ct (16 cols); K=128 -> 4 x wmma 16x16x32.
// A tile (16x128 f16, 4KB) staged once in LDS via async-copy (global_load_async_to_lds_b128)
// and shared by all 8 waves; rows padded 16B to spread LDS banks for the ds_load_b128s.
__global__ __launch_bounds__(256) void k_gemm(
    const half_t* __restrict__ xh, const half_t* __restrict__ Bp,
    const float* __restrict__ att_src, const float* __restrict__ att_dst,
    float* __restrict__ xp, float* __restrict__ a_src, float* __restrict__ a_dst, int Nn) {
    __shared__ half_t Atile[16 * 136];           // 136 = 128 + 8-half row pad
    const int t    = threadIdx.x;
    const int lane = t & 31;
    const int ct   = t >> 5;                     // head / column tile, 0..7
    const int node_base = blockIdx.x * 16;
    const int m    = lane & 15;
    const int hi4  = (lane & 16) ? 16 : 0;       // K sub-offset for half-wave

    // cooperative stage: 256 threads x 16B = whole tile in one shot
    {
        int r = t >> 4;                          // tile row 0..15
        int c = (t & 15) * 8;                    // halves
        int grow = node_base + r; if (grow >= Nn) grow = Nn - 1;
        const half_t* gp = xh + (size_t)grow * 128 + c;
        half_t* lp = &Atile[r * 136 + c];
#ifdef HAVE_ASYNC_LDS
        __builtin_amdgcn_global_load_async_to_lds_b128((AS1 v4i*)gp, (AS3 v4i*)lp, 0, 0);
#else
        *(v8h*)lp = *(const v8h*)gp;
#endif
    }
#ifdef HAVE_ASYNC_LDS
    wait_async0();
#endif
    __syncthreads();

    v8f acc = {};
    #pragma unroll
    for (int kc = 0; kc < 4; ++kc) {
        // A fragment (16x32 f16, contiguous k = 16h+s, same order as B pack): two contiguous 16B groups per lane, from LDS
        const half_t* ap = &Atile[m * 136 + kc * 32 + hi4];
        v8h alo = *(const v8h*)(ap);
        v8h ahi = *(const v8h*)(ap + 8);
        // B fragment: pre-packed, 32B contiguous per lane (L2-resident, 32KB total)
        const half_t* bpp = Bp + (((ct * 4 + kc) * 32 + lane) << 4);
        v8h blo = *(const v8h*)(bpp);
        v8h bhi = *(const v8h*)(bpp + 8);
        v16h a, b;
        #pragma unroll
        for (int i = 0; i < 8; ++i) { a[i] = alo[i]; a[i + 8] = ahi[i]; b[i] = blo[i]; b[i + 8] = bhi[i]; }
        acc = __builtin_amdgcn_wmma_f32_16x16x32_f16(false, a, false, b, (short)0, acc, false, false);
    }

    const int rbase = node_base + (hi4 ? 8 : 0); // C/D layout: vgpr r, lane -> row rbase+r
    const int col   = ct * 16 + m;

    // fused head reductions: a_src[n,h]=sum_c xp*att ; 16-lane butterfly per half-wave
    float cs = att_src[col], cd = att_dst[col];
    float ss[8], sd[8];
    #pragma unroll
    for (int r = 0; r < 8; ++r) { ss[r] = acc[r] * cs; sd[r] = acc[r] * cd; }
    #pragma unroll
    for (int mask = 1; mask < 16; mask <<= 1) {
        #pragma unroll
        for (int r = 0; r < 8; ++r) {
            ss[r] += __shfl_xor(ss[r], mask, 32);
            sd[r] += __shfl_xor(sd[r], mask, 32);
        }
    }

    if (node_base + 16 <= Nn) {                  // full tile (always true when 16 | N)
        #pragma unroll
        for (int r = 0; r < 8; ++r)
            xp[(size_t)(rbase + r) * 128 + col] = acc[r];
        if (m == 0) {
            #pragma unroll
            for (int r = 0; r < 8; ++r) {
                a_src[(size_t)(rbase + r) * 8 + ct] = ss[r];
                a_dst[(size_t)(rbase + r) * 8 + ct] = sd[r];
            }
        }
    } else {                                     // tail tile: clamped (duplicate-safe)
        #pragma unroll
        for (int r = 0; r < 8; ++r) {
            int rr = rbase + r; if (rr >= Nn) rr = Nn - 1;
            xp[(size_t)rr * 128 + col] = acc[r];
        }
        if (m == 0) {
            #pragma unroll
            for (int r = 0; r < 8; ++r) {
                int rr = rbase + r; if (rr >= Nn) rr = Nn - 1;
                a_src[(size_t)rr * 8 + ct] = ss[r];
                a_dst[(size_t)rr * 8 + ct] = sd[r];
            }
        }
    }
}

// ---------- 6. logits + leaky relu + segment max (encoded uint atomicMax) ----------
__global__ void k_alpha_max(const int* __restrict__ ei, const float* __restrict__ ea,
                            const float* __restrict__ la,
                            const float* __restrict__ a_src, const float* __restrict__ a_dst,
                            const float* __restrict__ kE,
                            float* __restrict__ exbuf, unsigned* __restrict__ mx, int E, int Nn) {
    int e = blockIdx.x * blockDim.x + threadIdx.x;
    int EN = E + Nn;
    if (e >= EN) return;
    int s, d; float w;
    if (e < E) { s = ei[e]; d = ei[E + e]; w = ea[e]; }
    else       { s = d = e - E;            w = la[e - E]; }
    float4 s0 = ((const float4*)(a_src + (size_t)s * 8))[0];
    float4 s1 = ((const float4*)(a_src + (size_t)s * 8))[1];
    float4 d0 = ((const float4*)(a_dst + (size_t)d * 8))[0];
    float4 d1 = ((const float4*)(a_dst + (size_t)d * 8))[1];
    float as[8] = {s0.x, s0.y, s0.z, s0.w, s1.x, s1.y, s1.z, s1.w};
    float ad[8] = {d0.x, d0.y, d0.z, d0.w, d1.x, d1.y, d1.z, d1.w};
    #pragma unroll
    for (int h = 0; h < 8; ++h) {
        float al = as[h] + ad[h] + w * kE[h];
        al = (al > 0.f) ? al : 0.2f * al;          // leaky relu 0.2
        exbuf[(size_t)e * 8 + h] = al;
        atomicMax(&mx[(size_t)d * 8 + h], fenc(al));
    }
}

// ---------- 7. exp(alpha - max) + segment sum ----------
__global__ void k_exp_denom(const int* __restrict__ ei, const unsigned* __restrict__ mx,
                            float* __restrict__ exbuf, float* __restrict__ denom, int E, int Nn) {
    int e = blockIdx.x * blockDim.x + threadIdx.x;
    int EN = E + Nn;
    if (e >= EN) return;
    int d = (e < E) ? ei[E + e] : (e - E);
    #pragma unroll
    for (int h = 0; h < 8; ++h) {
        float al = exbuf[(size_t)e * 8 + h];
        float ex = __expf(al - fdec(mx[(size_t)d * 8 + h]));
        exbuf[(size_t)e * 8 + h] = ex;
        atomicAdd(&denom[(size_t)d * 8 + h], ex);
    }
}

// ---------- 8. aggregate: one wave per edge, 512B coalesced gather + f32 atomic scatter ----------
__global__ void k_aggregate(const int* __restrict__ ei, const float* __restrict__ exbuf,
                            const float* __restrict__ denom, const float* __restrict__ xp,
                            float* __restrict__ out, int E, int Nn) {
    long long gt = (long long)blockIdx.x * blockDim.x + threadIdx.x;
    int lane = threadIdx.x & 31;
    long long e = gt >> 5;
    int EN = E + Nn;
    if (e >= EN) return;
    int s, d;
    if (e < E) { s = ei[e]; d = ei[E + (int)e]; }
    else       { s = d = (int)e - E; }
    int h = lane >> 2;                              // 4 lanes per head (16 cols)
    float ex  = exbuf[(size_t)e * 8 + h];
    float den = denom[(size_t)d * 8 + h];
    float attn = ex / (den + 1e-16f);
    float4 xv = ((const float4*)(xp + (size_t)s * 128))[lane];
    float* op = out + (size_t)d * 128 + lane * 4;
    atomicAdd(op + 0, attn * xv.x);
    atomicAdd(op + 1, attn * xv.y);
    atomicAdd(op + 2, attn * xv.z);
    atomicAdd(op + 3, attn * xv.w);
}

// ---------- 9. bias + residual + layernorm + relu (one wave per row, in-place on d_out) ----------
__global__ __launch_bounds__(256) void k_final(const float* __restrict__ x,
                                               const float* __restrict__ bias,
                                               const float* __restrict__ gamma,
                                               const float* __restrict__ beta,
                                               float* __restrict__ out, int Nn) {
    int lane = threadIdx.x & 31;
    int wv   = threadIdx.x >> 5;
    int row  = blockIdx.x * 8 + wv;
    if (row >= Nn) return;
    float* orow = out + (size_t)row * 128;
    float4 ov = ((float4*)orow)[lane];
    float4 xv = ((const float4*)(x + (size_t)row * 128))[lane];
    float4 bv = ((const float4*)bias)[lane];
    float h0 = ov.x + xv.x + bv.x;
    float h1 = ov.y + xv.y + bv.y;
    float h2 = ov.z + xv.z + bv.z;
    float h3 = ov.w + xv.w + bv.w;
    float sum = h0 + h1 + h2 + h3;
    float sq  = h0 * h0 + h1 * h1 + h2 * h2 + h3 * h3;
    #pragma unroll
    for (int mask = 1; mask < 32; mask <<= 1) {
        sum += __shfl_xor(sum, mask, 32);
        sq  += __shfl_xor(sq,  mask, 32);
    }
    float mu  = sum * (1.f / 128.f);
    float var = sq * (1.f / 128.f) - mu * mu;
    float rs  = rsqrtf(var + 1e-5f);
    float4 gv = ((const float4*)gamma)[lane];
    float4 tv = ((const float4*)beta)[lane];
    float r0 = (h0 - mu) * rs * gv.x + tv.x; r0 = r0 > 0.f ? r0 : 0.f;
    float r1 = (h1 - mu) * rs * gv.y + tv.y; r1 = r1 > 0.f ? r1 : 0.f;
    float r2 = (h2 - mu) * rs * gv.z + tv.z; r2 = r2 > 0.f ? r2 : 0.f;
    float r3 = (h3 - mu) * rs * gv.w + tv.w; r3 = r3 > 0.f ? r3 : 0.f;
    ((float4*)orow)[lane] = make_float4(r0, r1, r2, r3);
}

extern "C" void kernel_launch(void* const* d_in, const int* in_sizes, int n_in,
                              void* d_out, int out_size, void* d_ws, size_t ws_size,
                              hipStream_t stream) {
    const float* x        = (const float*)d_in[0];
    const int*   ei       = (const int*)d_in[1];     // edge_index [2,E]
    const float* ea       = (const float*)d_in[2];
    const float* W        = (const float*)d_in[3];
    const float* att_src  = (const float*)d_in[4];
    const float* att_dst  = (const float*)d_in[5];
    const float* att_edge = (const float*)d_in[6];
    const float* W_edge   = (const float*)d_in[7];
    const float* bias     = (const float*)d_in[8];
    const float* gamma    = (const float*)d_in[9];
    const float* beta     = (const float*)d_in[10];
    const int Nn = in_sizes[0] / 128;
    const int E  = in_sizes[2];
    const int EN = E + Nn;

    // workspace carve-out (256B aligned)
    char* wsb = (char*)d_ws;
    size_t off = 0;
    auto carve = [&](size_t bytes) -> char* {
        char* p = wsb + off;
        off = (off + bytes + 255) & ~(size_t)255;
        return p;
    };
    half_t*   xh    = (half_t*)  carve((size_t)Nn * 128 * 2);
    half_t*   Bp    = (half_t*)  carve(128 * 128 * 2);
    float*    xp    = (float*)   carve((size_t)Nn * 128 * 4);
    float*    aS    = (float*)   carve((size_t)Nn * 8 * 4);
    float*    aD    = (float*)   carve((size_t)Nn * 8 * 4);
    unsigned* cnt   = (unsigned*)carve((size_t)Nn * 4);
    float*    ssum  = (float*)   carve((size_t)Nn * 4);
    float*    la    = (float*)   carve((size_t)Nn * 4);
    float*    kE    = (float*)   carve(32);
    unsigned* mx    = (unsigned*)carve((size_t)Nn * 8 * 4);
    float*    den   = (float*)   carve((size_t)Nn * 8 * 4);
    float*    exb   = (float*)   carve((size_t)EN * 8 * 4);

    (void)hipMemsetAsync(cnt,   0, (size_t)Nn * 4, stream);
    (void)hipMemsetAsync(ssum,  0, (size_t)Nn * 4, stream);
    (void)hipMemsetAsync(mx,    0, (size_t)Nn * 8 * 4, stream);   // 0 == encoded "-infinity"
    (void)hipMemsetAsync(den,   0, (size_t)Nn * 8 * 4, stream);
    (void)hipMemsetAsync(d_out, 0, (size_t)Nn * 128 * 4, stream); // out accumulator

    k_edge_coef<<<1, 32, 0, stream>>>(W_edge, att_edge, kE);
    k_loop_stats<<<(E + 255) / 256, 256, 0, stream>>>(ei, ea, cnt, ssum, E);
    k_loop_attr<<<(Nn + 255) / 256, 256, 0, stream>>>(cnt, ssum, la, Nn);
    k_convert_x<<<(Nn * 128 + 255) / 256, 256, 0, stream>>>(x, xh, Nn * 128);
    k_pack_w<<<(16384 + 255) / 256, 256, 0, stream>>>(W, Bp);
    k_gemm<<<(Nn + 15) / 16, 256, 0, stream>>>(xh, Bp, att_src, att_dst, xp, aS, aD, Nn);
    k_alpha_max<<<(EN + 255) / 256, 256, 0, stream>>>(ei, ea, la, aS, aD, kE, exb, mx, E, Nn);
    k_exp_denom<<<(EN + 255) / 256, 256, 0, stream>>>(ei, mx, exb, den, E, Nn);
    long long aggT = (long long)EN * 32;
    k_aggregate<<<(unsigned)((aggT + 255) / 256), 256, 0, stream>>>(ei, exb, den, xp, (float*)d_out, E, Nn);
    k_final<<<(Nn + 7) / 8, 256, 0, stream>>>(x, bias, gamma, beta, (float*)d_out, Nn);
}
